// Model_42700564857442
// MI455X (gfx1250) — matrix operand order repaired and checked by reading, not compiled, not run on hardware
//
#include <hip/hip_runtime.h>
#include <hip/hip_bf16.h>

// ---------------------------------------------------------------------------
// GraphSAGE link predictor for MI455X (gfx1250, wave32, WMMA).
//
// GEMMs run on v_wmma_f32_16x16x32_bf16 (bf16 in, f32 accumulate).
// Weights are staged in LDS in *fragment order* so a B fragment is one
// aligned 32-byte LDS read (2x ds_load_b128) per lane.
// Edge aggregation = float4 gathers + native global_atomic_add_f32
// (relaxed, agent scope). All node-feature tables are 25.6MB, so the whole
// irregular working set lives in the 192MB L2. Mean division is folded into
// the WMMA A-fragment load as a per-row scale (1/max(deg,1)).
// ---------------------------------------------------------------------------

typedef __attribute__((ext_vector_type(16))) __bf16 v16bf;
typedef __attribute__((ext_vector_type(8)))  float  v8f;

#define N_NODES 50000
#define CFEAT   128
#define XFEAT   384

__device__ __forceinline__ v8f wmma_bf16(v16bf a, v16bf b, v8f c) {
  // 8 args: (neg_a, A, neg_b, B, c_mod, C, reuse_a, reuse_b)
  return __builtin_amdgcn_wmma_f32_16x16x32_bf16(
      false, a, false, b, (short)0, c, false, false);
}

__device__ __forceinline__ void atomic_fadd(float* p, float v) {
  __hip_atomic_fetch_add(p, v, __ATOMIC_RELAXED, __HIP_MEMORY_SCOPE_AGENT);
}

// A-fragment: 16x32 bf16 tile of row-major f32 matrix, in contiguous K order (same as B):
//   row = lane&15; elem i -> K = kb+i (i = 0..15),
//   kb = k0 + (lane>=16 ? 16 : 0).  Four contiguous float4 loads.
__device__ __forceinline__ v16bf load_a_frag(const float* __restrict__ A,
                                             int lda, int m0, int k0,
                                             int lane, float scale) {
  const int row = lane & 15;
  const int kb  = k0 + ((lane & 16) ? 16 : 0);
  const float* p = A + (size_t)(m0 + row) * lda + kb;
  float4 f0 = *(const float4*)(p + 0);
  float4 f1 = *(const float4*)(p + 4);
  float4 f2 = *(const float4*)(p + 8);
  float4 f3 = *(const float4*)(p + 12);
  v16bf a;
  a[0]  = (__bf16)(f0.x * scale); a[1]  = (__bf16)(f0.y * scale);
  a[2]  = (__bf16)(f0.z * scale); a[3]  = (__bf16)(f0.w * scale);
  a[4]  = (__bf16)(f1.x * scale); a[5]  = (__bf16)(f1.y * scale);
  a[6]  = (__bf16)(f1.z * scale); a[7]  = (__bf16)(f1.w * scale);
  a[8]  = (__bf16)(f2.x * scale); a[9]  = (__bf16)(f2.y * scale);
  a[10] = (__bf16)(f2.z * scale); a[11] = (__bf16)(f2.w * scale);
  a[12] = (__bf16)(f3.x * scale); a[13] = (__bf16)(f3.y * scale);
  a[14] = (__bf16)(f3.z * scale); a[15] = (__bf16)(f3.w * scale);
  return a;
}

// LDS fragment-order index for weight element W[k][n] (n fastest in global):
//   idx(k,n) = ((k/16)*128 + n)*16 + (k%16)
// => the 16 K-values one lane needs for a B fragment (col fixed, K contiguous
//    within a 16-run, matching the A-fragment K order) are one aligned 32B run.
__device__ __forceinline__ int wfrag_idx(int k, int n) {
  return (((k >> 4) << 7) + n) * 16 + (k & 15);
}

// B-fragment: 32x16 tile; col = lane&15, elem i -> K = k0+(lane>=16?16:0)+i.
// One 32-byte vector read -> 2x ds_load_b128.
__device__ __forceinline__ v16bf load_b_frag_lds(const __bf16* __restrict__ w,
                                                 int n0, int k0, int lane) {
  const int col = lane & 15;
  const int kb  = k0 + ((lane & 16) ? 16 : 0);
  return *(const v16bf*)(w + (size_t)(((kb >> 4) << 7) + n0 + col) * 16);
}

// ---------------------------------------------------------------------------
// GEMM0: out[M,128] = A[M,K] @ W[K,128] + bias.   grid.x = M/80, block = 256.
// Each block: 5 M-tiles x 8 N-tiles; wave w owns N-tile w, loops 5 M-tiles,
// reusing each B fragment across 5 WMMAs. W staged in LDS (bf16, frag order).
// ---------------------------------------------------------------------------
__global__ void __launch_bounds__(256)
gemm_bias_kernel(const float* __restrict__ A, const float* __restrict__ W,
                 const float* __restrict__ bias, float* __restrict__ out,
                 int K) {
  extern __shared__ __bf16 smem[];              // K*128 bf16, frag order
  const int tid = threadIdx.x;
  for (int i = tid; i < K * CFEAT; i += 256) {
    const int k = i >> 7, n = i & 127;
    smem[wfrag_idx(k, n)] = (__bf16)W[i];
  }
  __syncthreads();

  const int wave = tid >> 5, lane = tid & 31;
  const int m0 = blockIdx.x * 80;
  const int n0 = wave * 16;

  v8f acc[5];
#pragma unroll
  for (int t = 0; t < 5; ++t) acc[t] = (v8f){};

  for (int k0 = 0; k0 < K; k0 += 32) {
    v16bf b = load_b_frag_lds(smem, n0, k0, lane);
#pragma unroll
    for (int t = 0; t < 5; ++t) {
      v16bf a = load_a_frag(A, K, m0 + t * 16, k0, lane, 1.0f);
      acc[t] = wmma_bf16(a, b, acc[t]);
    }
  }

  const int col = lane & 15;
  const int rb  = (lane & 16) ? 8 : 0;
  const float bv = bias[n0 + col];
#pragma unroll
  for (int t = 0; t < 5; ++t)
#pragma unroll
    for (int r = 0; r < 8; ++r)
      out[(size_t)(m0 + t * 16 + rb + r) * CFEAT + n0 + col] = acc[t][r] + bv;
}

// ---------------------------------------------------------------------------
// SAGE GEMM: out = act( (agg*rdeg) @ WL + H @ WR + bias ), all [M,128],K=128.
// ---------------------------------------------------------------------------
__global__ void __launch_bounds__(256)
sage_gemm_kernel(const float* __restrict__ AGG, const float* __restrict__ H,
                 const float* __restrict__ RDEG, const float* __restrict__ WL,
                 const float* __restrict__ WR, const float* __restrict__ bias,
                 float* __restrict__ out, int do_relu) {
  extern __shared__ __bf16 smem[];              // 2 * 128*128 bf16 = 64KB
  __bf16* lwl = smem;
  __bf16* lwr = smem + CFEAT * CFEAT;
  const int tid = threadIdx.x;
  for (int i = tid; i < CFEAT * CFEAT; i += 256) {
    const int k = i >> 7, n = i & 127;
    const int j = wfrag_idx(k, n);
    lwl[j] = (__bf16)WL[i];
    lwr[j] = (__bf16)WR[i];
  }
  __syncthreads();

  const int wave = tid >> 5, lane = tid & 31;
  const int m0 = blockIdx.x * 80;
  const int n0 = wave * 16;

  float sc[5];
#pragma unroll
  for (int t = 0; t < 5; ++t) sc[t] = RDEG[m0 + t * 16 + (lane & 15)];

  v8f acc[5];
#pragma unroll
  for (int t = 0; t < 5; ++t) acc[t] = (v8f){};

  // mean-aggregated neighbors @ WL
  for (int k0 = 0; k0 < CFEAT; k0 += 32) {
    v16bf b = load_b_frag_lds(lwl, n0, k0, lane);
#pragma unroll
    for (int t = 0; t < 5; ++t) {
      v16bf a = load_a_frag(AGG, CFEAT, m0 + t * 16, k0, lane, sc[t]);
      acc[t] = wmma_bf16(a, b, acc[t]);
    }
  }
  // self features @ WR
  for (int k0 = 0; k0 < CFEAT; k0 += 32) {
    v16bf b = load_b_frag_lds(lwr, n0, k0, lane);
#pragma unroll
    for (int t = 0; t < 5; ++t) {
      v16bf a = load_a_frag(H, CFEAT, m0 + t * 16, k0, lane, 1.0f);
      acc[t] = wmma_bf16(a, b, acc[t]);
    }
  }

  const int col = lane & 15;
  const int rb  = (lane & 16) ? 8 : 0;
  const float bv = bias[n0 + col];
#pragma unroll
  for (int t = 0; t < 5; ++t)
#pragma unroll
    for (int r = 0; r < 8; ++r) {
      float v = acc[t][r] + bv;
      if (do_relu) v = fmaxf(v, 0.0f);
      out[(size_t)(m0 + t * 16 + rb + r) * CFEAT + n0 + col] = v;
    }
}

// ---------------------------------------------------------------------------
// Zero a float buffer (float4 stores).
// ---------------------------------------------------------------------------
__global__ void zero_kernel(float* __restrict__ p, long n4) {
  long i = (long)blockIdx.x * blockDim.x + threadIdx.x;
  long stride = (long)gridDim.x * blockDim.x;
  float4 z = make_float4(0.f, 0.f, 0.f, 0.f);
  for (; i < n4; i += stride) ((float4*)p)[i] = z;
}

// ---------------------------------------------------------------------------
// Degree count + reciprocal.
// ---------------------------------------------------------------------------
__global__ void deg_kernel(const int* __restrict__ ei, float* __restrict__ deg,
                           int E) {
  int i = blockIdx.x * blockDim.x + threadIdx.x;
  if (i < E) atomic_fadd(&deg[ei[E + i]], 1.0f);   // dst row of edge_index
}

__global__ void rdeg_kernel(const float* __restrict__ deg,
                            float* __restrict__ rdeg, int n) {
  int i = blockIdx.x * blockDim.x + threadIdx.x;
  if (i < n) rdeg[i] = 1.0f / fmaxf(deg[i], 1.0f);
}

// ---------------------------------------------------------------------------
// Scatter-add: agg[dst] += H[src].  One wave per edge; each lane moves a
// float4 (128 floats / 32 lanes = 4 per lane), 4 native f32 atomics.
// ---------------------------------------------------------------------------
__global__ void scatter_kernel(const float* __restrict__ H,
                               const int* __restrict__ ei,
                               float* __restrict__ agg, int E) {
  int e = blockIdx.x * (blockDim.x >> 5) + (threadIdx.x >> 5);
  if (e >= E) return;
  const int lane = threadIdx.x & 31;
  const int s = ei[e];
  const int d = ei[E + e];
  float4 v = *(const float4*)(H + (size_t)s * CFEAT + lane * 4);
  float* p = agg + (size_t)d * CFEAT + lane * 4;
  atomic_fadd(p + 0, v.x);
  atomic_fadd(p + 1, v.y);
  atomic_fadd(p + 2, v.z);
  atomic_fadd(p + 3, v.w);
}

// ---------------------------------------------------------------------------
// Classifier: out[e] = dot(h[head], h[tail]).  One wave per label edge,
// float4 loads, wave32 xor-shuffle reduction.
// ---------------------------------------------------------------------------
__global__ void classify_kernel(const float* __restrict__ H,
                                const int* __restrict__ eli,
                                float* __restrict__ out, int EL) {
  int e = blockIdx.x * (blockDim.x >> 5) + (threadIdx.x >> 5);
  if (e >= EL) return;
  const int lane = threadIdx.x & 31;
  const int hrow = eli[e];
  const int trow = eli[EL + e];
  float4 a = *(const float4*)(H + (size_t)hrow * CFEAT + lane * 4);
  float4 b = *(const float4*)(H + (size_t)trow * CFEAT + lane * 4);
  float s = a.x * b.x + a.y * b.y + a.z * b.z + a.w * b.w;
#pragma unroll
  for (int off = 16; off > 0; off >>= 1) s += __shfl_xor(s, off, 32);
  if (lane == 0) out[e] = s;
}

// ---------------------------------------------------------------------------
extern "C" void kernel_launch(void* const* d_in, const int* in_sizes, int n_in,
                              void* d_out, int out_size, void* d_ws,
                              size_t ws_size, hipStream_t stream) {
  const float* x     = (const float*)d_in[0];   // [50000,384]
  const int*   ei    = (const int*)d_in[1];     // [2,E] flat: src then dst
  const int*   eli   = (const int*)d_in[2];     // [2,EL]
  const float* W_lin = (const float*)d_in[3];   // [384,128]
  const float* b_lin = (const float*)d_in[4];   // [128]
  const float* W1_l  = (const float*)d_in[5];
  const float* b1    = (const float*)d_in[6];
  const float* W1_r  = (const float*)d_in[7];
  const float* W2_l  = (const float*)d_in[8];
  const float* b2    = (const float*)d_in[9];
  const float* W2_r  = (const float*)d_in[10];

  const int E  = in_sizes[1] / 2;
  const int EL = in_sizes[2] / 2;

  const size_t NF = (size_t)N_NODES * CFEAT;    // 6.4M floats per table
  float* ws   = (float*)d_ws;
  float* h0   = ws;                              // [N,128]
  float* agg  = h0 + NF;                         // [N,128]
  float* h1   = agg + NF;                        // [N,128]
  float* h2   = h1 + NF;                         // [N,128]
  float* deg  = h2 + NF;                         // [N]
  float* rdeg = deg + N_NODES;                   // [N]
  (void)ws_size; (void)n_in; (void)out_size;

  const int MBLK = 625;                          // 50000 / 80
  const size_t shm0 = (size_t)XFEAT * CFEAT * sizeof(__bf16);     // 96KB
  const size_t shm1 = (size_t)2 * CFEAT * CFEAT * sizeof(__bf16); // 64KB

  // degree (shared by both layers)
  zero_kernel<<<256, 256, 0, stream>>>(deg, (long)N_NODES / 4);
  deg_kernel<<<(E + 255) / 256, 256, 0, stream>>>(ei, deg, E);
  rdeg_kernel<<<(N_NODES + 255) / 256, 256, 0, stream>>>(deg, rdeg, N_NODES);

  // h0 = x @ W_lin + b_lin
  gemm_bias_kernel<<<MBLK, 256, shm0, stream>>>(x, W_lin, b_lin, h0, XFEAT);

  // layer 1
  zero_kernel<<<2048, 256, 0, stream>>>(agg, (long)(NF / 4));
  scatter_kernel<<<(E + 7) / 8, 256, 0, stream>>>(h0, ei, agg, E);
  sage_gemm_kernel<<<MBLK, 256, shm1, stream>>>(agg, h0, rdeg, W1_l, W1_r, b1,
                                                h1, /*relu=*/1);

  // layer 2
  zero_kernel<<<2048, 256, 0, stream>>>(agg, (long)(NF / 4));
  scatter_kernel<<<(E + 7) / 8, 256, 0, stream>>>(h1, ei, agg, E);
  sage_gemm_kernel<<<MBLK, 256, shm1, stream>>>(agg, h1, rdeg, W2_l, W2_r, b2,
                                                h2, /*relu=*/0);

  // classifier
  classify_kernel<<<(EL + 7) / 8, 256, 0, stream>>>(h2, eli, (float*)d_out, EL);
}
